// Network_15874199126374
// MI455X (gfx1250) — matrix operand order repaired and checked by reading, not compiled, not run on hardware
//
#include <hip/hip_runtime.h>
#include <math.h>

typedef __attribute__((ext_vector_type(16))) __bf16 v16bf;
typedef __attribute__((ext_vector_type(8)))  float  v8f;

#define D_DIM 1024
#define BATCH 4096
#define NOPS  5
#define BM 64
#define BN 64
#define BK 32

struct LayerDesc {
  const float* X0;      // first input segment [4096,1024] fp32
  const float* X1;      // second segment (cat edge) or nullptr
  const float* W0;      // weight base for segment 0 (rows 0..1023 of [NOPS,K,D])
  const float* W1;      // weight base for segment 1 (rows 1024..2047) or nullptr
  long opStride;        // K_total * D  (elements between op-k blocks)
  int  l;               // layer index (bias / genotype row)
  int  pad;
};

struct NodeArgs {
  LayerDesc layers[6];
  int n_layers;
  int pad;
  const float* genotype; // [18,5]
  const float* bs;       // [18,5,1024]
  float* out;            // [4096,1024]
};

// pack two fp32 -> packed bf16 pair (round-to-nearest-even), lo in [15:0]
__device__ __forceinline__ unsigned pack_bf16(float lo, float hi) {
  unsigned ulo = __float_as_uint(lo);
  ulo += 0x7FFFu + ((ulo >> 16) & 1u);
  unsigned uhi = __float_as_uint(hi);
  uhi += 0x7FFFu + ((uhi >> 16) & 1u);
  return (ulo >> 16) | (uhi & 0xFFFF0000u);
}

union FragU { v16bf v; unsigned u[8]; };

__global__ __launch_bounds__(256) void darts_node_kernel(NodeArgs args) {
  // ping-pong staging: 2 * (4 KB A + 20 KB B) = 48 KB of the 320 KB WGP LDS
  __shared__ __align__(16) unsigned short As [2][BM][BK];
  __shared__ __align__(16) unsigned short BsT[2][NOPS][BN][BK];

  const int tid    = threadIdx.x;
  const int lane   = tid & 31;
  const int wave   = tid >> 5;        // 0..7
  const int wm     = wave & 3;        // row block of wave
  const int wn     = wave >> 2;       // col block of wave
  const int bm0    = blockIdx.x * BM;
  const int bn0    = blockIdx.y * BN;
  const int l15    = lane & 15;
  const int hiHalf = lane >> 4;       // 0 or 1

  // A staging: thread owns 8 consecutive floats of one row
  const int ar  = tid >> 2;
  const int ac0 = (tid & 3) * 8;
  // B staging: thread owns an 8-deep K-column of one output column
  const int bc  = tid & 63;
  const int brb = (tid >> 6) * 8;

  v8f out_acc[2];
  #pragma unroll
  for (int t = 0; t < 2; ++t)
    #pragma unroll
    for (int v = 0; v < 8; ++v) out_acc[t][v] = 0.0f;

  for (int L = 0; L < args.n_layers; ++L) {
    const LayerDesc ld = args.layers[L];

    v8f acc[NOPS][2];
    #pragma unroll
    for (int k = 0; k < NOPS; ++k)
      #pragma unroll
      for (int t = 0; t < 2; ++t)
        #pragma unroll
        for (int v = 0; v < 8; ++v) acc[k][t][v] = 0.0f;

    const int nseg   = (ld.X1 != nullptr) ? 2 : 1;
    const int nsteps = nseg << 5;          // 32 K-steps per 1024-wide segment

    unsigned a_pk[4];
    unsigned b_pk[NOPS][4];

    // fetch tile for K-step `step` into packed registers
    auto gload = [&](int step) {
      const int s  = step >> 5;
      const int kk = (step & 31) * BK;
      const float* X = s ? ld.X1 : ld.X0;
      const float* W = s ? ld.W1 : ld.W0;
      // A: 8 consecutive floats of row (bm0+ar), cols kk+ac0..+7
      {
        const float* src = X + (long)(bm0 + ar) * D_DIM + kk + ac0;
        const float4 f0 = *(const float4*)src;
        const float4 f1 = *(const float4*)(src + 4);
        a_pk[0] = pack_bf16(f0.x, f0.y);
        a_pk[1] = pack_bf16(f0.z, f0.w);
        a_pk[2] = pack_bf16(f1.x, f1.y);
        a_pk[3] = pack_bf16(f1.z, f1.w);
      }
      // B: per op, 8 floats down column (bn0+bc), rows kk+brb..+7 (lane-coalesced)
      #pragma unroll
      for (int k = 0; k < NOPS; ++k) {
        const float* src = W + (long)k * ld.opStride
                             + (long)(kk + brb) * D_DIM + bn0 + bc;
        float f[8];
        #pragma unroll
        for (int j = 0; j < 8; ++j) f[j] = src[(long)j * D_DIM];
        b_pk[k][0] = pack_bf16(f[0], f[1]);
        b_pk[k][1] = pack_bf16(f[2], f[3]);
        b_pk[k][2] = pack_bf16(f[4], f[5]);
        b_pk[k][3] = pack_bf16(f[6], f[7]);
      }
    };

    gload(0);
    int buf = 0;

    for (int step = 0; step < nsteps; ++step) {
      // ---- commit staged tile to LDS[buf]: one b128 store each ----
      *(uint4*)&As[buf][ar][ac0] =
          make_uint4(a_pk[0], a_pk[1], a_pk[2], a_pk[3]);
      #pragma unroll
      for (int k = 0; k < NOPS; ++k)
        *(uint4*)&BsT[buf][k][bc][brb] =
            make_uint4(b_pk[k][0], b_pk[k][1], b_pk[k][2], b_pk[k][3]);

      __syncthreads();   // single barrier per K-step (ping-pong makes it safe)

      // ---- issue next tile's global loads so latency hides under WMMA ----
      if (step + 1 < nsteps) gload(step + 1);

      // ---- A fragment: lane = M row, contiguous K halves, k = 16*hiHalf + s (same order as B) ----
      FragU au;
      {
        const int m  = wm * 16 + l15;
        const int kb = hiHalf * 16;
        #pragma unroll
        for (int j = 0; j < 8; ++j) {
          const int K = 2 * j + kb;
          au.u[j] = *(const unsigned*)&As[buf][m][K];
        }
      }

      // ---- 10 WMMAs with double-buffered B fragments: frag(kt+1) loads
      //      are issued before wmma(kt) so LDS latency hides under the XDL op
      const int kb2 = hiHalf * 16;
      FragU bu[2];
      {
        const int n = wn * 32 + 0 * 16 + l15;   // kt = 0 -> k=0, t=0
        #pragma unroll
        for (int j = 0; j < 8; ++j)
          bu[0].u[j] = *(const unsigned*)&BsT[buf][0][n][kb2 + 2 * j];
      }
      #pragma unroll
      for (int kt = 0; kt < NOPS * 2; ++kt) {
        const int k = kt >> 1;
        const int t = kt & 1;
        if (kt + 1 < NOPS * 2) {
          const int nk = (kt + 1) >> 1;
          const int nt = (kt + 1) & 1;
          const int n  = wn * 32 + nt * 16 + l15;
          #pragma unroll
          for (int j = 0; j < 8; ++j)
            bu[(kt + 1) & 1].u[j] =
                *(const unsigned*)&BsT[buf][nk][n][kb2 + 2 * j];
        }
        acc[k][t] = __builtin_amdgcn_wmma_f32_16x16x32_bf16(
            false, au.v, false, bu[kt & 1].v, (short)0, acc[k][t], false, false);
      }
      buf ^= 1;
    }

    // ---- epilogue: bias + activation mix weighted by genotype ----
    const float* gen  = args.genotype + (long)ld.l * NOPS;
    const float* brow = args.bs + (long)ld.l * NOPS * D_DIM;
    #pragma unroll
    for (int k = 0; k < NOPS; ++k) {
      const float g = gen[k];
      #pragma unroll
      for (int t = 0; t < 2; ++t) {
        const int n = bn0 + wn * 32 + t * 16 + l15;
        const float bias = brow[k * D_DIM + n];
        #pragma unroll
        for (int v = 0; v < 8; ++v) {
          const float h = acc[k][t][v] + bias;
          float f;
          if      (k == 0) f = h;
          else if (k == 1) f = h > 0.0f ? h : 0.0f;
          else if (k == 2) f = tanhf(h);
          else if (k == 3) f = 1.0f / (1.0f + expf(-h));
          else             f = h > 0.0f ? h : 0.2f * h;
          out_acc[t][v] += g * f;
        }
      }
    }
  }

  // ---- store node output (fp32), C/D layout: VGPR v -> M = v + 8*hiHalf ----
  #pragma unroll
  for (int t = 0; t < 2; ++t) {
    const int n = bn0 + wn * 32 + t * 16 + l15;
    #pragma unroll
    for (int v = 0; v < 8; ++v) {
      const int m = bm0 + wm * 16 + v + hiHalf * 8;
      args.out[(long)m * D_DIM + n] = out_acc[t][v];
    }
  }
}

extern "C" void kernel_launch(void* const* d_in, const int* in_sizes, int n_in,
                              void* d_out, int out_size, void* d_ws, size_t ws_size,
                              hipStream_t stream) {
  const float* s0       = (const float*)d_in[0];
  const float* s1       = (const float*)d_in[1];
  const float* genotype = (const float*)d_in[2];
  const float* Ws       = (const float*)d_in[3];
  const float* Wb       = (const float*)d_in[4];
  const float* bs       = (const float*)d_in[5];
  float* out = (float*)d_out;

  const long SZ = (long)BATCH * D_DIM;
  float* n0 = (float*)d_ws;
  float* n1 = n0 + SZ;
  float* n2 = n1 + SZ;

  const long WS_BLK = 5L * 1024 * 1024;   // one Ws[si] block
  const long WB_BLK = 5L * 2048 * 1024;   // one Wb[bi] block
  const long WS_OP  = 1024L * 1024;
  const long WB_OP  = 2048L * 1024;

  auto mkS = [&](int si, int l, const float* X) {
    LayerDesc d{};
    d.X0 = X; d.X1 = nullptr;
    d.W0 = Ws + si * WS_BLK; d.W1 = nullptr;
    d.opStride = WS_OP; d.l = l;
    return d;
  };
  auto mkB = [&](int bi, int l) {
    LayerDesc d{};
    d.X0 = s0; d.X1 = s1;
    d.W0 = Wb + bi * WB_BLK;
    d.W1 = Wb + bi * WB_BLK + 1024L * 1024;   // rows 1024..2047
    d.opStride = WB_OP; d.l = l;
    return d;
  };

  dim3 grid(BATCH / BM, D_DIM / BN);
  dim3 block(256);

  NodeArgs a{};
  a.genotype = genotype;
  a.bs = bs;

  // node 0 -> n0
  a.layers[0] = mkS(0, 0, s0);
  a.layers[1] = mkS(1, 1, s1);
  a.layers[2] = mkB(0, 2);
  a.n_layers = 3; a.out = n0;
  hipLaunchKernelGGL(darts_node_kernel, grid, block, 0, stream, a);

  // node 1 -> n1
  a.layers[0] = mkS(2, 3, s0);
  a.layers[1] = mkS(3, 4, s1);
  a.layers[2] = mkB(1, 5);
  a.layers[3] = mkS(4, 6, n0);
  a.n_layers = 4; a.out = n1;
  hipLaunchKernelGGL(darts_node_kernel, grid, block, 0, stream, a);

  // node 2 -> n2
  a.layers[0] = mkS(5, 7, s0);
  a.layers[1] = mkS(6, 8, s1);
  a.layers[2] = mkB(2, 9);
  a.layers[3] = mkS(7, 10, n0);
  a.layers[4] = mkS(8, 11, n1);
  a.n_layers = 5; a.out = n2;
  hipLaunchKernelGGL(darts_node_kernel, grid, block, 0, stream, a);

  // node 3 -> final output
  a.layers[0] = mkS(9, 12, s0);
  a.layers[1] = mkS(10, 13, s1);
  a.layers[2] = mkB(3, 14);
  a.layers[3] = mkS(11, 15, n0);
  a.layers[4] = mkS(12, 16, n1);
  a.layers[5] = mkS(13, 17, n2);
  a.n_layers = 6; a.out = out;
  hipLaunchKernelGGL(darts_node_kernel, grid, block, 0, stream, a);
}
